// CustomMHA_78743930405302
// MI455X (gfx1250) — matrix operand order repaired and checked by reading, not compiled, not run on hardware
//
#include <hip/hip_runtime.h>
#include <hip/hip_bf16.h>

typedef __bf16 bf16;
typedef __attribute__((ext_vector_type(16))) __bf16 v16bf;
typedef __attribute__((ext_vector_type(8)))  __bf16 v8bf;
typedef __attribute__((ext_vector_type(8)))  float  v8f;

#define B_  512
#define L_  64
#define E_  1024
#define H_  16
#define HD_ 64
#define DM_ 1024
#define SMOL_CH 32
#define SMOL_HID 256
#define SMOL_GEN 256

__device__ __forceinline__ int imin(int a, int b) { return a < b ? a : b; }

__device__ __forceinline__ v16bf load_frag16(const bf16* p0) {
  // A-fragment layout: per lane two 8-element chunks at K+0 and K+16
  v8bf a1 = *(const v8bf*)p0;
  v8bf a2 = *(const v8bf*)(p0 + 16);
  return __builtin_shufflevector(a1, a2, 0,1,2,3,4,5,6,7,8,9,10,11,12,13,14,15);
}

// ---------------------------------------------------------------------------
// f32 -> bf16 flat convert
// ---------------------------------------------------------------------------
__global__ void cvt_f32_bf16(const float* __restrict__ in, bf16* __restrict__ out, long n) {
  long i = (long)blockIdx.x * blockDim.x + threadIdx.x;
  long stride = (long)gridDim.x * blockDim.x;
  for (; i < n; i += stride) out[i] = (bf16)in[i];
}

// ---------------------------------------------------------------------------
// W[K,N] f32  ->  Wt[N,K] bf16   (transpose + convert; one-time, tiny cost)
// ---------------------------------------------------------------------------
__global__ void transpose_cvt(const float* __restrict__ W, bf16* __restrict__ Wt, int K, int N) {
  long total = (long)K * N;
  long i = (long)blockIdx.x * blockDim.x + threadIdx.x;
  long stride = (long)gridDim.x * blockDim.x;
  for (; i < total; i += stride) {
    int k = (int)(i / N);
    int n = (int)(i - (long)k * N);
    Wt[(long)n * K + k] = (bf16)W[i];
  }
}

// ---------------------------------------------------------------------------
// Generic WMMA bf16 GEMM:  C[M,N] = A[M,K] @ Bt[N,K]^T  (+bias) (SiLU?)
// Block: 256 threads = 8 waves arranged 4(M) x 2(N).
// Each wave: 32x64 register tile = 2 A-frags x 4 B-frags = 8 WMMA / K-step.
// Block tile: 128 x 128.   flags: bit0 = bf16 out, bit1 = SiLU epilogue
// ---------------------------------------------------------------------------
__global__ __launch_bounds__(256) void gemm_bf16_wmma(
    const bf16* __restrict__ A, const bf16* __restrict__ Bt,
    const float* __restrict__ bias, float* __restrict__ Cf, bf16* __restrict__ Cb,
    int M, int N, int K, int flags)
{
  const int lane = threadIdx.x & 31;
  const int wave = threadIdx.x >> 5;
  const int wm   = wave & 3;
  const int wn   = wave >> 2;
  const int lr   = lane & 15;
  const int hi   = lane >> 4;
  const int m0   = blockIdx.y * 128 + wm * 32;
  const int n0   = blockIdx.x * 128 + wn * 64;

  v8f acc0[4] = {};
  v8f acc1[4] = {};

  const bf16* arow0 = A + (size_t)imin(m0 + lr,      M - 1) * K + hi * 8;
  const bf16* arow1 = A + (size_t)imin(m0 + 16 + lr, M - 1) * K + hi * 8;
  const bf16* brow[4];
  int nidx[4];
#pragma unroll
  for (int t = 0; t < 4; ++t) {
    nidx[t] = n0 + t * 16 + lr;
    brow[t] = Bt + (size_t)imin(nidx[t], N - 1) * K + hi * 8;
  }

  for (int k0 = 0; k0 < K; k0 += 32) {
    // WGP-scope software prefetch a few K-steps ahead (global_prefetch_b8,
    // locality 3 -> pull into all cache levels, stop on WGP-cache hit)
    __builtin_prefetch(arow0 + k0 + 256, 0, 3);
    __builtin_prefetch(arow1 + k0 + 256, 0, 3);
    __builtin_prefetch(brow[0] + k0 + 256, 0, 3);

    v16bf af0 = load_frag16(arow0 + k0);
    v16bf af1 = load_frag16(arow1 + k0);
#pragma unroll
    for (int t = 0; t < 4; ++t) {
      v16bf bfrag = load_frag16(brow[t] + k0);
      acc0[t] = __builtin_amdgcn_wmma_f32_16x16x32_bf16(
          false, af0, false, bfrag, (short)0, acc0[t], false, false);
      acc1[t] = __builtin_amdgcn_wmma_f32_16x16x32_bf16(
          false, af1, false, bfrag, (short)0, acc1[t], false, false);
    }
  }

#pragma unroll
  for (int t = 0; t < 4; ++t) {
    const int n = nidx[t];
#pragma unroll
    for (int r = 0; r < 8; ++r) {
#pragma unroll
      for (int u = 0; u < 2; ++u) {
        const int m = m0 + u * 16 + r + 8 * hi;
        if (m < M && n < N) {
          float v = u ? acc1[t][r] : acc0[t][r];
          if (bias) v += bias[n];
          if (flags & 2) v = v / (1.0f + __expf(-v));   // SiLU
          if (flags & 1) Cb[(size_t)m * N + n] = (bf16)v;
          else           Cf[(size_t)m * N + n] = v;
        }
      }
    }
  }
}

// ---------------------------------------------------------------------------
// Row LayerNorm: X[rows,C] f32 -> Y[rows,C] bf16 ; one block per row
// ---------------------------------------------------------------------------
__global__ __launch_bounds__(256) void layernorm_rows(
    const float* __restrict__ X, const float* __restrict__ g,
    const float* __restrict__ bta, bf16* __restrict__ Y, int C)
{
  __shared__ float red[256];
  const int tid = threadIdx.x;
  const float* x = X + (size_t)blockIdx.x * C;
  bf16* y = Y + (size_t)blockIdx.x * C;

  float s = 0.f;
  for (int i = tid; i < C; i += 256) s += x[i];
  red[tid] = s; __syncthreads();
  for (int off = 128; off; off >>= 1) {
    if (tid < off) red[tid] += red[tid + off];
    __syncthreads();
  }
  const float mean = red[0] / (float)C;
  __syncthreads();

  float v = 0.f;
  for (int i = tid; i < C; i += 256) { float d = x[i] - mean; v += d * d; }
  red[tid] = v; __syncthreads();
  for (int off = 128; off; off >>= 1) {
    if (tid < off) red[tid] += red[tid + off];
    __syncthreads();
  }
  const float inv = rsqrtf(red[0] / (float)C + 1e-3f);
  for (int i = tid; i < C; i += 256)
    y[i] = (bf16)((x[i] - mean) * inv * g[i] + bta[i]);
}

// ---------------------------------------------------------------------------
// Fused attention per (b,h): S = QK^T/8 + smol ; softmax ; O = P V
// block = 128 threads (4 waves); wave w owns query rows [16w, 16w+16)
// Q,K,V are bf16 [B*L, DM] with head h at column h*64.
// ---------------------------------------------------------------------------
__global__ __launch_bounds__(128) void attn_fused(
    const bf16* __restrict__ Q, const bf16* __restrict__ Kb,
    const bf16* __restrict__ V, const float* __restrict__ smol,
    bf16* __restrict__ O)
{
  __shared__ bf16 qs[64 * 64];
  __shared__ bf16 ks[64 * 64];
  __shared__ bf16 vts[64 * 64];   // V transposed: vts[d][n]
  __shared__ bf16 ps[64 * 64];    // softmax probs, bf16

  const int bh = blockIdx.x;
  const int b  = bh >> 4;
  const int h  = bh & 15;
  const int tid  = threadIdx.x;
  const int lane = tid & 31;
  const int wave = tid >> 5;
  const size_t base = ((size_t)b * L_) * DM_ + (size_t)h * HD_;

  // stage Q, K, V (V transposed) into LDS
  for (int idx = tid; idx < 64 * 8; idx += 128) {
    const int row = idx >> 3;
    const int c   = (idx & 7) * 8;
    v8bf qv = *(const v8bf*)(Q  + base + (size_t)row * DM_ + c);
    v8bf kv = *(const v8bf*)(Kb + base + (size_t)row * DM_ + c);
    v8bf vv = *(const v8bf*)(V  + base + (size_t)row * DM_ + c);
    *(v8bf*)(qs + row * 64 + c) = qv;
    *(v8bf*)(ks + row * 64 + c) = kv;
#pragma unroll
    for (int i = 0; i < 8; ++i) vts[(c + i) * 64 + row] = vv[i];
  }
  __syncthreads();

  const int lr = lane & 15;
  const int hi = lane >> 4;
  const int m0 = wave * 16;

  // ---- S = Q K^T ----
  v8f s[4] = {};
#pragma unroll
  for (int k0 = 0; k0 < 64; k0 += 32) {
    v16bf af = load_frag16(qs + (m0 + lr) * 64 + k0 + hi * 8);
#pragma unroll
    for (int t = 0; t < 4; ++t) {
      v16bf bfrag = load_frag16(ks + (t * 16 + lr) * 64 + k0 + hi * 8);
      s[t] = __builtin_amdgcn_wmma_f32_16x16x32_bf16(
          false, af, false, bfrag, (short)0, s[t], false, false);
    }
  }

  // ---- scale + smolgen bias + row softmax ----
  const float scale = 0.125f;                 // 1/sqrt(64)
  const float* sm = smol + (size_t)bh * (L_ * L_);
#pragma unroll
  for (int r = 0; r < 8; ++r) {
    const int m = m0 + r + 8 * hi;            // C/D layout: lanes16-31 -> M+8
    float mx = -1e30f;
#pragma unroll
    for (int t = 0; t < 4; ++t) {
      const int n = t * 16 + lr;
      float v = s[t][r] * scale + sm[m * 64 + n];
      s[t][r] = v;
      mx = fmaxf(mx, v);
    }
    // reduce across the 16 lanes that hold this row (masks stay within halves)
#pragma unroll
    for (int msk = 1; msk < 16; msk <<= 1) mx = fmaxf(mx, __shfl_xor(mx, msk, 32));
    float sum = 0.f;
#pragma unroll
    for (int t = 0; t < 4; ++t) {
      float e = __expf(s[t][r] - mx);
      s[t][r] = e;
      sum += e;
    }
#pragma unroll
    for (int msk = 1; msk < 16; msk <<= 1) sum += __shfl_xor(sum, msk, 32);
    const float inv = 1.0f / sum;
#pragma unroll
    for (int t = 0; t < 4; ++t)
      ps[m * 64 + t * 16 + lr] = (bf16)(s[t][r] * inv);
  }
  __syncthreads();

  // ---- O = P V  (A = P rows, B[k][d] = V[k][d] = vts[d][k]) ----
  v8f o[4] = {};
#pragma unroll
  for (int k0 = 0; k0 < 64; k0 += 32) {
    v16bf af = load_frag16(ps + (m0 + lr) * 64 + k0 + hi * 8);
#pragma unroll
    for (int t = 0; t < 4; ++t) {
      v16bf bfrag = load_frag16(vts + (t * 16 + lr) * 64 + k0 + hi * 8);
      o[t] = __builtin_amdgcn_wmma_f32_16x16x32_bf16(
          false, af, false, bfrag, (short)0, o[t], false, false);
    }
  }

  // store O as bf16 into attn_out[B*L, DM] at column h*64
#pragma unroll
  for (int t = 0; t < 4; ++t) {
    const int d = t * 16 + lr;
#pragma unroll
    for (int r = 0; r < 8; ++r) {
      const int m = m0 + r + 8 * hi;
      O[base + (size_t)m * DM_ + d] = (bf16)o[t][r];
    }
  }
}

// ---------------------------------------------------------------------------
// host-side orchestration
// ---------------------------------------------------------------------------
extern "C" void kernel_launch(void* const* d_in, const int* in_sizes, int n_in,
                              void* d_out, int out_size, void* d_ws, size_t ws_size,
                              hipStream_t stream)
{
  (void)in_sizes; (void)n_in; (void)out_size; (void)ws_size;
  const float* x     = (const float*)d_in[0];
  const float* wq    = (const float*)d_in[1];
  const float* bq    = (const float*)d_in[2];
  const float* wk    = (const float*)d_in[3];
  const float* bk    = (const float*)d_in[4];
  const float* wv    = (const float*)d_in[5];
  const float* bv    = (const float*)d_in[6];
  const float* wo    = (const float*)d_in[7];
  const float* bo    = (const float*)d_in[8];
  const float* sc_w  = (const float*)d_in[9];
  const float* sh1_w = (const float*)d_in[10];
  const float* sh1_b = (const float*)d_in[11];
  const float* ln1_g = (const float*)d_in[12];
  const float* ln1_b = (const float*)d_in[13];
  const float* sg_w  = (const float*)d_in[14];
  const float* sg_b  = (const float*)d_in[15];
  const float* ln2_g = (const float*)d_in[16];
  const float* ln2_b = (const float*)d_in[17];
  const float* swg_w = (const float*)d_in[18];
  float* out = (float*)d_out;

  char* p = (char*)d_ws;
  auto take = [&](size_t bytes) -> char* {
    char* r = p;
    p += (bytes + 255) & ~(size_t)255;
    return r;
  };
  const size_t TOK = (size_t)B_ * L_;                 // 32768
  bf16* xb    = (bf16*)take(TOK * E_ * 2);
  bf16* qb    = (bf16*)take(TOK * DM_ * 2);
  bf16* kb    = (bf16*)take(TOK * DM_ * 2);
  bf16* vb    = (bf16*)take(TOK * DM_ * 2);
  bf16* aout  = (bf16*)take(TOK * DM_ * 2);
  float* smol = (float*)take((size_t)B_ * H_ * L_ * L_ * 4);
  bf16* wqt   = (bf16*)take((size_t)DM_ * E_ * 2);
  bf16* wkt   = (bf16*)take((size_t)DM_ * E_ * 2);
  bf16* wvt   = (bf16*)take((size_t)DM_ * E_ * 2);
  bf16* wot   = (bf16*)take((size_t)E_ * DM_ * 2);
  bf16* sct   = (bf16*)take((size_t)SMOL_CH * E_ * 2);
  bf16* sh1t  = (bf16*)take((size_t)SMOL_HID * (L_ * SMOL_CH) * 2);
  bf16* sgt   = (bf16*)take((size_t)(H_ * SMOL_GEN) * SMOL_HID * 2);
  bf16* swgt  = (bf16*)take((size_t)(L_ * L_) * SMOL_GEN * 2);
  bf16* comp  = (bf16*)take(TOK * SMOL_CH * 2);
  float* hpre = (float*)take((size_t)B_ * SMOL_HID * 4);
  bf16* hid   = (bf16*)take((size_t)B_ * SMOL_HID * 2);
  float* gpre = (float*)take((size_t)B_ * H_ * SMOL_GEN * 4);
  bf16* gen   = (bf16*)take((size_t)B_ * H_ * SMOL_GEN * 2);

  // --- conversions ---
  cvt_f32_bf16<<<4096, 256, 0, stream>>>(x, xb, (long)TOK * E_);
  transpose_cvt<<<2048, 256, 0, stream>>>(wq, wqt, E_, DM_);
  transpose_cvt<<<2048, 256, 0, stream>>>(wk, wkt, E_, DM_);
  transpose_cvt<<<2048, 256, 0, stream>>>(wv, wvt, E_, DM_);
  transpose_cvt<<<2048, 256, 0, stream>>>(wo, wot, DM_, E_);
  transpose_cvt<<<256,  256, 0, stream>>>(sc_w, sct, E_, SMOL_CH);
  transpose_cvt<<<1024, 256, 0, stream>>>(sh1_w, sh1t, L_ * SMOL_CH, SMOL_HID);
  transpose_cvt<<<2048, 256, 0, stream>>>(sg_w, sgt, SMOL_HID, H_ * SMOL_GEN);
  transpose_cvt<<<2048, 256, 0, stream>>>(swg_w, swgt, SMOL_GEN, L_ * L_);

  // --- Q/K/V projections (bf16 out, f32 accumulate) ---
  {
    dim3 g(DM_ / 128, (int)(TOK / 128));
    gemm_bf16_wmma<<<g, 256, 0, stream>>>(xb, wqt, bq, nullptr, qb, (int)TOK, DM_, E_, 1);
    gemm_bf16_wmma<<<g, 256, 0, stream>>>(xb, wkt, bk, nullptr, kb, (int)TOK, DM_, E_, 1);
    gemm_bf16_wmma<<<g, 256, 0, stream>>>(xb, wvt, bv, nullptr, vb, (int)TOK, DM_, E_, 1);
  }

  // --- smolgen chain ---
  {
    dim3 g(1, (int)(TOK / 128));   // N=32 fits in one tile column
    gemm_bf16_wmma<<<g, 256, 0, stream>>>(xb, sct, nullptr, nullptr, comp,
                                          (int)TOK, SMOL_CH, E_, 1);
  }
  {
    dim3 g((SMOL_HID + 127) / 128, B_ / 128);  // [512,2048]@[2048,256], SiLU, f32
    gemm_bf16_wmma<<<g, 256, 0, stream>>>(comp, sh1t, sh1_b, hpre, nullptr,
                                          B_, SMOL_HID, L_ * SMOL_CH, 2);
  }
  layernorm_rows<<<B_, 256, 0, stream>>>(hpre, ln1_g, ln1_b, hid, SMOL_HID);
  {
    dim3 g((H_ * SMOL_GEN) / 128, B_ / 128);   // [512,256]@[256,4096], SiLU, f32
    gemm_bf16_wmma<<<g, 256, 0, stream>>>(hid, sgt, sg_b, gpre, nullptr,
                                          B_, H_ * SMOL_GEN, SMOL_HID, 2);
  }
  layernorm_rows<<<B_, 256, 0, stream>>>(gpre, ln2_g, ln2_b, gen, H_ * SMOL_GEN);
  {
    dim3 g((L_ * L_) / 128, (B_ * H_) / 128);  // [8192,256]@[256,4096] -> smol f32
    gemm_bf16_wmma<<<g, 256, 0, stream>>>(gen, swgt, nullptr, smol, nullptr,
                                          B_ * H_, L_ * L_, SMOL_GEN, 0);
  }

  // --- fused attention: one workgroup per (b,h) ---
  attn_fused<<<B_ * H_, 128, 0, stream>>>(qb, kb, vb, smol, aout);

  // --- output projection -> f32 d_out ---
  {
    dim3 g(E_ / 128, (int)(TOK / 128));
    gemm_bf16_wmma<<<g, 256, 0, stream>>>(aout, wot, bo, out, nullptr,
                                          (int)TOK, E_, DM_, 0);
  }
}
